// DecoderWithLoss_59407987638334
// MI455X (gfx1250) — compile-verified, matrix operand order checked by reading, not run on hardware
//
#include <hip/hip_runtime.h>
#include <hip/hip_bf16.h>

// ---------------- problem constants (from reference) ----------------
#define BB   16
#define SYq  128
#define SXk  256
#define Hd   768
#define NHh  12
#define FFd  3072
#define Vv   30522
#define Lc   4
#define HDd  64
#define NHWc 2
#define SCALE_ATTN 0.125f   // 1/sqrt(64)

typedef __attribute__((ext_vector_type(16))) __bf16 v16bf;
typedef __attribute__((ext_vector_type(8)))  __bf16 v8bf;
typedef __attribute__((ext_vector_type(8)))  float  v8f;

// ---------------- small device helpers ----------------
__device__ __forceinline__ float wred_sum(float v) {
    #pragma unroll
    for (int o = 16; o > 0; o >>= 1) v += __shfl_xor(v, o, 32);
    return v;
}
__device__ __forceinline__ float wred_max(float v) {
    #pragma unroll
    for (int o = 16; o > 0; o >>= 1) v = fmaxf(v, __shfl_xor(v, o, 32));
    return v;
}
// guaranteed single-instruction pack: (lo,hi) -> {bf16(lo) | bf16(hi)<<16}
__device__ __forceinline__ unsigned pack_bf16(float lo, float hi) {
    unsigned r;
    asm("v_cvt_pk_bf16_f32 %0, %1, %2" : "=v"(r) : "v"(lo), "v"(hi));
    return r;
}

template<int ACT>
__device__ __forceinline__ float act_fn(float x) {
    if (ACT == 1) return 0.5f * x * (1.0f + erff(x * 0.70710678118654752f)); // exact gelu
    if (ACT == 2) return 1.0f / (1.0f + expf(-x));                          // sigmoid
    if (ACT == 3) return tanhf(x);                                          // tanh
    return x;
}

// ---------------- batched strided WMMA GEMM ----------------
// C[m,n] = act(sum_k A[m,k]*B[k,n] + bias[n])
// BT==0: B element (k,n) at Bm[k*ldb + n]   (row-major K x N; ldb even)
// BT==1: B element (k,n) at Bm[n*ldb + k]   (B given as N x K; 16B-aligned rows)
// Requirements (all call sites satisfy): K % 32 == 0, M % 64 == 0, N even.
// batch z: offX = (z/div)*soX + (z%div)*siX
// Block tile: 64 (M) x 128 (N) x 32 (K); 8 waves, each owns a 32x32 subtile
// (2x2 fragments => 4 WMMAs per wave per K-step). LDS double-buffered:
// one barrier per K-step; next tile is staged while current tile is computed.
#define LDT 40   // LDS row stride in bf16 elements (80B, 16B aligned, conflict-padded)

template<int BT, int ACT>
__global__ __launch_bounds__(256)
void gemm_wmma_kernel(const float* __restrict__ A, const float* __restrict__ Bm,
                      const float* __restrict__ bias, float* __restrict__ C,
                      int M, int N, int K, int lda, int ldb, int ldc, int div,
                      long soA, long siA, long soB, long siB, long soC, long siC)
{
    __shared__ __attribute__((aligned(16))) __bf16 As[2 * 64  * LDT];
    __shared__ __attribute__((aligned(16))) __bf16 Bs[2 * 128 * LDT];

    const int tid  = threadIdx.x;
    const int m0   = blockIdx.y * 64;
    const int n0   = blockIdx.x * 128;
    const int z    = blockIdx.z;
    const int zb   = z / div, zi = z % div;
    A  += zb * soA + (long)zi * siA;
    Bm += zb * soB + (long)zi * siB;
    C  += zb * soC + (long)zi * siC;

    const int w    = tid >> 5;
    const int lane = tid & 31;
    const int wm   = (w & 1) * 32;   // 2 waves along M, 32 rows each
    const int wn   = (w >> 1) * 32;  // 4 waves along N, 32 cols each
    const int lm   = lane & 15;
    const int hh   = lane >> 4;      // K-half selector for 16-bit fragments

    v8f acc00 = {}, acc01 = {}, acc10 = {}, acc11 = {};

    // staging coordinates
    const int arow = tid >> 2, acol = (tid & 3) * 8;    // A: 64 rows x 32 k
    const int bn2  = (tid & 63) * 2, bkp = tid >> 6;    // BT==0: col pair + k-pair base
    const int brow = tid >> 1, bcol = (tid & 1) * 16;   // BT==1: 128 rows x 32 k

    // branch-free staging: clamp addresses (stay aligned & in-bounds), zero by select
    auto stage = [&](int k0, __bf16* Asb, __bf16* Bsb) {
        {   // ---- A (64x32) ----
            const int  gmr = m0 + arow;
            const bool okA = (gmr < M);
            const float* ap = A + (long)(okA ? gmr : 0) * lda + k0 + acol;
            const float4 f0 = *(const float4*)ap;
            const float4 f1 = *(const float4*)(ap + 4);
            unsigned u0 = pack_bf16(f0.x, f0.y), u1 = pack_bf16(f0.z, f0.w);
            unsigned u2 = pack_bf16(f1.x, f1.y), u3 = pack_bf16(f1.z, f1.w);
            if (!okA) { u0 = 0u; u1 = 0u; u2 = 0u; u3 = 0u; }
            *(uint4*)&Asb[arow * LDT + acol] = make_uint4(u0, u1, u2, u3);
        }
        if (BT == 0) {   // ---- B row-major KxN -> Bs[n][k] (128x32) ----
            const int  gn = n0 + bn2;
            const bool ok = (gn + 1 < N);          // N even => pairwise validity
            const int  cc = ok ? gn : 0;           // clamped column (8B aligned)
            #pragma unroll
            for (int i = 0; i < 4; ++i) {
                const int kpe = bkp + 4 * i;       // k-pair 0..15
                const float* r0 = Bm + (long)(k0 + 2 * kpe) * ldb + cc;
                const float2 fa = *(const float2*)r0;
                const float2 fb = *(const float2*)(r0 + ldb);
                unsigned uA = pack_bf16(fa.x, fb.x);
                unsigned uB = pack_bf16(fa.y, fb.y);
                if (!ok) { uA = 0u; uB = 0u; }
                *(unsigned*)&Bsb[(bn2    ) * LDT + 2 * kpe] = uA;
                *(unsigned*)&Bsb[(bn2 + 1) * LDT + 2 * kpe] = uB;
            }
        } else {         // ---- B as NxK -> Bs[n][k] (128x32) ----
            const int  gnr = n0 + brow;
            const bool okB = (gnr < N);
            const float* bp = Bm + (long)(okB ? gnr : 0) * ldb + k0 + bcol;
            const float4 f0 = *(const float4*)bp;
            const float4 f1 = *(const float4*)(bp + 4);
            const float4 f2 = *(const float4*)(bp + 8);
            const float4 f3 = *(const float4*)(bp + 12);
            unsigned u0 = pack_bf16(f0.x, f0.y), u1 = pack_bf16(f0.z, f0.w);
            unsigned u2 = pack_bf16(f1.x, f1.y), u3 = pack_bf16(f1.z, f1.w);
            unsigned u4 = pack_bf16(f2.x, f2.y), u5 = pack_bf16(f2.z, f2.w);
            unsigned u6 = pack_bf16(f3.x, f3.y), u7 = pack_bf16(f3.z, f3.w);
            if (!okB) { u0=0u;u1=0u;u2=0u;u3=0u;u4=0u;u5=0u;u6=0u;u7=0u; }
            __bf16* dst = &Bsb[brow * LDT + bcol];
            *(uint4*)dst       = make_uint4(u0, u1, u2, u3);
            *(uint4*)(dst + 8) = make_uint4(u4, u5, u6, u7);
        }
    };

    stage(0, As, Bs);
    int p = 0;
    for (int k0 = 0; k0 < K; k0 += 32) {
        __syncthreads();   // staged tile visible; previous readers done
        if (k0 + 32 < K)
            stage(k0 + 32, p ? As : As + 64 * LDT, p ? Bs : Bs + 128 * LDT);
        const __bf16* cA = p ? As + 64 * LDT : As;
        const __bf16* cB = p ? Bs + 128 * LDT : Bs;

        // fragments: 16-bit layout = two contiguous 8-elem K runs per lane
        union UF { v16bf v; v8bf h[2]; } a0, a1, b0, b1;
        const int ko = 8 * hh;
        a0.h[0] = *(const v8bf*)&cA[(wm + lm)      * LDT + ko];
        a0.h[1] = *(const v8bf*)&cA[(wm + lm)      * LDT + 16 + ko];
        a1.h[0] = *(const v8bf*)&cA[(wm + 16 + lm) * LDT + ko];
        a1.h[1] = *(const v8bf*)&cA[(wm + 16 + lm) * LDT + 16 + ko];
        b0.h[0] = *(const v8bf*)&cB[(wn + lm)      * LDT + ko];
        b0.h[1] = *(const v8bf*)&cB[(wn + lm)      * LDT + 16 + ko];
        b1.h[0] = *(const v8bf*)&cB[(wn + 16 + lm) * LDT + ko];
        b1.h[1] = *(const v8bf*)&cB[(wn + 16 + lm) * LDT + 16 + ko];

        acc00 = __builtin_amdgcn_wmma_f32_16x16x32_bf16(false, a0.v, false, b0.v,
                                                        (short)0, acc00, false, false);
        acc01 = __builtin_amdgcn_wmma_f32_16x16x32_bf16(false, a0.v, false, b1.v,
                                                        (short)0, acc01, false, false);
        acc10 = __builtin_amdgcn_wmma_f32_16x16x32_bf16(false, a1.v, false, b0.v,
                                                        (short)0, acc10, false, false);
        acc11 = __builtin_amdgcn_wmma_f32_16x16x32_bf16(false, a1.v, false, b1.v,
                                                        (short)0, acc11, false, false);
        p ^= 1;
    }

    // ---- epilogue (M % 64 == 0: no row guards). C/D layout: lane = col,
    //      VGPR v = row v + 8*hh within the 16x16 tile. ----
    #pragma unroll
    for (int ni = 0; ni < 2; ++ni) {
        const int col = n0 + wn + 16 * ni + lm;
        if (col < N) {
            const float bv = bias ? bias[col] : 0.0f;
            const v8f ac0 = ni ? acc01 : acc00;
            const v8f ac1 = ni ? acc11 : acc10;
            const long r0 = m0 + wm + 8 * hh;
            #pragma unroll
            for (int v = 0; v < 8; ++v) {
                C[(r0 + v)      * (long)ldc + col] = act_fn<ACT>(ac0[v] + bv);
                C[(r0 + 16 + v) * (long)ldc + col] = act_fn<ACT>(ac1[v] + bv);
            }
        }
    }
}

// ---------------- masked scaled softmax (wave per row) ----------------
// rows = B*NH*Sq ; row r: b = (r/Sq)/NH, q = r%Sq ; mask row stride == Sk here
__global__ __launch_bounds__(256)
void attn_softmax_kernel(float* __restrict__ S, const int* __restrict__ mask,
                         int Sq, int Sk, int causal)
{
    const int w = threadIdx.x >> 5, lane = threadIdx.x & 31;
    const int row = blockIdx.x * 8 + w;
    const int q = row % Sq;
    const int b = (row / Sq) / NHh;
    float* p = S + (long)row * Sk;
    const int* mrow = mask + b * Sk;

    const int cnt = Sk >> 5;   // 4 or 8
    float vals[8];
    float mx = -1e30f;
    #pragma unroll 8
    for (int i = 0; i < 8; ++i) {
        if (i < cnt) {
            const int k = lane + i * 32;
            float v = p[k] * SCALE_ATTN;
            const bool ok = (mrow[k] != 0) && (!causal || (k <= q));
            v += ok ? 0.0f : -1e9f;
            vals[i] = v;
            mx = fmaxf(mx, v);
        }
    }
    mx = wred_max(mx);
    float s = 0.0f;
    #pragma unroll 8
    for (int i = 0; i < 8; ++i) {
        if (i < cnt) { vals[i] = expf(vals[i] - mx); s += vals[i]; }
    }
    s = wred_sum(s);
    const float inv = 1.0f / s;
    #pragma unroll 8
    for (int i = 0; i < 8; ++i)
        if (i < cnt) p[lane + i * 32] = vals[i] * inv;
}

// ---------------- LayerNorm over H=768 (wave per row), optional residual ----------------
__global__ __launch_bounds__(256)
void ln_kernel(const float* __restrict__ X, const float* __restrict__ R,
               const float* __restrict__ g, const float* __restrict__ be,
               float* __restrict__ O, int rows)
{
    const int w = threadIdx.x >> 5, lane = threadIdx.x & 31;
    const int row = blockIdx.x * 8 + w;
    if (row >= rows) return;
    const long base = (long)row * Hd;
    float v[24];
    float s = 0.0f;
    #pragma unroll
    for (int i = 0; i < 24; ++i) {
        const int c = lane + i * 32;
        float x = X[base + c];
        if (R) x += R[base + c];
        v[i] = x; s += x;
    }
    const float mean = wred_sum(s) * (1.0f / Hd);
    float var = 0.0f;
    #pragma unroll
    for (int i = 0; i < 24; ++i) { const float d = v[i] - mean; var += d * d; }
    var = wred_sum(var) * (1.0f / Hd);
    const float inv = rsqrtf(var + 1e-12f);
    #pragma unroll
    for (int i = 0; i < 24; ++i) {
        const int c = lane + i * 32;
        O[base + c] = (v[i] - mean) * inv * g[c] + be[c];
    }
}

// ---------------- embedding gather + positional add ----------------
__global__ __launch_bounds__(256)
void emb_kernel(const int* __restrict__ y, const float* __restrict__ we,
                const float* __restrict__ pe, float* __restrict__ O)
{
    const long i = (long)blockIdx.x * 256 + threadIdx.x;  // [0, 2048*768)
    const int row = (int)(i / Hd), c = (int)(i % Hd);
    const int tok = y[row];
    const int s = row % SYq;
    O[i] = we[(long)tok * Hd + c] + pe[(long)s * Hd + c];
}

// ---------------- highway combine: h = g*n + (1-g)*h ----------------
__global__ __launch_bounds__(256)
void hw_combine_kernel(const float* __restrict__ g, const float* __restrict__ n,
                       float* __restrict__ h)
{
    const long i = (long)blockIdx.x * 256 + threadIdx.x;
    const float gg = g[i];
    h[i] = gg * n[i] + (1.0f - gg) * h[i];
}

// ---------------- in-place log_softmax (block per row, any length) ----------------
__global__ __launch_bounds__(256)
void logsoftmax_kernel(float* __restrict__ X, int cols)
{
    __shared__ float red[8];
    float* x = X + (long)blockIdx.x * cols;
    const int t = threadIdx.x, lane = t & 31, wv = t >> 5;

    float mx = -1e30f;
    for (int c = t; c < cols; c += 256) mx = fmaxf(mx, x[c]);
    mx = wred_max(mx);
    if (lane == 0) red[wv] = mx;
    __syncthreads();
    if (wv == 0) {
        float m2 = (lane < 8) ? red[lane] : -1e30f;
        m2 = wred_max(m2);
        if (lane == 0) red[0] = m2;
    }
    __syncthreads();
    const float bmax = red[0];
    __syncthreads();

    float s = 0.0f;
    for (int c = t; c < cols; c += 256) s += expf(x[c] - bmax);
    s = wred_sum(s);
    if (lane == 0) red[wv] = s;
    __syncthreads();
    if (wv == 0) {
        float s2 = (lane < 8) ? red[lane] : 0.0f;
        s2 = wred_sum(s2);
        if (lane == 0) red[0] = s2;
    }
    __syncthreads();
    const float lse = bmax + logf(red[0]);
    for (int c = t; c < cols; c += 256) x[c] = x[c] - lse;
}

// ---------------- host-side GEMM dispatcher ----------------
static void gemm(hipStream_t st, int BT, int ACT,
                 const float* A, const float* Bm, const float* bias, float* C,
                 int M, int N, int K, int lda, int ldb, int ldc,
                 int batch, int div,
                 long soA, long siA, long soB, long siB, long soC, long siC)
{
    dim3 g((N + 127) / 128, (M + 63) / 64, batch), b(256, 1, 1);
#define GCALL(bt, act) gemm_wmma_kernel<bt, act><<<g, b, 0, st>>>( \
        A, Bm, bias, C, M, N, K, lda, ldb, ldc, div, soA, siA, soB, siB, soC, siC)
    if (BT == 0) {
        switch (ACT) {
            case 0: GCALL(0, 0); break;
            case 1: GCALL(0, 1); break;
            case 2: GCALL(0, 2); break;
            case 3: GCALL(0, 3); break;
        }
    } else {
        GCALL(1, 0);
    }
#undef GCALL
}

// ============================================================================
// Input flattening assumption (setup_inputs dict order, nested dicts in order):
//  0 y[int]  1 y_mask[int]  2 x_reps  3 x_mask[int]
//  4 word_emb 5 pos_emb 6 emb_ln_s 7 emb_ln_b
//  8 qw 9 qb 10 kw 11 kb 12 vw 13 vb 14 ow 15 ob 16 ln1s 17 ln1b
// 18 cqw 19 cqb 20 ckw 21 ckb 22 cvw 23 cvb 24 cow 25 cob 26 ln2s 27 ln2b
// 28 f1w 29 f1b 30 f2w 31 f2b 32 ln3s 33 ln3b        (each with leading L=4)
// 34 lm_dw 35 lm_db 36 lm_ln_s 37 lm_ln_b 38 lm_dec_w 39 lm_dec_b
// 40..43 hw_b {gw,gb,nw,nb}  44..47 hw_e {gw,gb,nw,nb}
// 48 ts_sw 49 ts_sb 50 ts_ew 51 ts_eb
// Outputs: concat(lm_log, span_b_log, span_e_log, ts_s_log, ts_e_log).
// Note: pad_bias adds a per-row constant before log_softmax -> cancels; skipped.
// ============================================================================
extern "C" void kernel_launch(void* const* d_in, const int* in_sizes, int n_in,
                              void* d_out, int out_size, void* d_ws, size_t ws_size,
                              hipStream_t stream)
{
    (void)in_sizes; (void)n_in; (void)out_size; (void)ws_size;
    auto F = [&](int i) { return (const float*)d_in[i]; };
    const int* y      = (const int*)d_in[0];
    const int* y_mask = (const int*)d_in[1];
    const float* x_reps = F(2);
    const int* x_mask = (const int*)d_in[3];

    const long M2 = (long)BB * SYq;          // 2048
    const long MX = (long)BB * SXk;          // 4096
    const long HH = (long)Hd * Hd;           // 589824

    // workspace layout (floats)
    float* ws = (float*)d_ws;
    size_t off = 0;
    auto alloc = [&](size_t n) { float* p = ws + off; off += n; return p; };
    float* h    = alloc(M2 * Hd);            // hidden state / y_rep
    float* qb_  = alloc(M2 * Hd);            // Q projections
    float* kb_  = alloc(MX * Hd);            // K projections (max 4096 rows)
    float* vb_  = alloc(MX * Hd);            // V projections
    float* big  = alloc((size_t)BB * NHh * SYq * SXk); // scores (6.29M) == FFN mid size
    float* c1   = alloc(M2 * Hd);            // ctx / t
    float* c2   = alloc(M2 * Hd);            // attn-out / gate / proj
    float* c3   = alloc(M2 * Hd);            // nlin
    float* hy   = alloc(M2 * Hd);            // highway running state

    // output regions
    float* out    = (float*)d_out;
    float* out_lm = out;
    float* out_sb = out_lm + M2 * (long)Vv;
    float* out_se = out_sb + M2 * (long)SXk;
    float* out_ts = out_se + M2 * (long)SXk;
    float* out_te = out_ts + M2 * (long)SXk;

    const dim3 b256(256, 1, 1);
    const int lnBlocks   = (int)(M2 / 8);            // 256
    const int elemBlocks = (int)(M2 * Hd / 256);     // 6144
    const int smBlocksS  = BB * NHh * SYq / 8;       // 3072

    // ---- embedding + LN ----
    emb_kernel<<<elemBlocks, b256, 0, stream>>>(y, F(4), F(5), c2);
    ln_kernel<<<lnBlocks, b256, 0, stream>>>(c2, nullptr, F(6), F(7), h, (int)M2);

    // ---- transformer layers ----
    for (int l = 0; l < Lc; ++l) {
        const float* qw = F(8)  + (size_t)l * HH; const float* qbias = F(9)  + (size_t)l * Hd;
        const float* kw = F(10) + (size_t)l * HH; const float* kbias = F(11) + (size_t)l * Hd;
        const float* vw = F(12) + (size_t)l * HH; const float* vbias = F(13) + (size_t)l * Hd;
        const float* ow = F(14) + (size_t)l * HH; const float* obias = F(15) + (size_t)l * Hd;
        const float* l1s = F(16) + (size_t)l * Hd; const float* l1b = F(17) + (size_t)l * Hd;
        const float* cqw = F(18) + (size_t)l * HH; const float* cqb = F(19) + (size_t)l * Hd;
        const float* ckw = F(20) + (size_t)l * HH; const float* ckb = F(21) + (size_t)l * Hd;
        const float* cvw = F(22) + (size_t)l * HH; const float* cvb = F(23) + (size_t)l * Hd;
        const float* cow = F(24) + (size_t)l * HH; const float* cob = F(25) + (size_t)l * Hd;
        const float* l2s = F(26) + (size_t)l * Hd; const float* l2b = F(27) + (size_t)l * Hd;
        const float* f1w = F(28) + (size_t)l * Hd * FFd; const float* f1b = F(29) + (size_t)l * FFd;
        const float* f2w = F(30) + (size_t)l * FFd * Hd; const float* f2b = F(31) + (size_t)l * Hd;
        const float* l3s = F(32) + (size_t)l * Hd; const float* l3b = F(33) + (size_t)l * Hd;

        // --- self attention ---
        gemm(stream, 0, 0, h, qw, qbias, qb_, (int)M2, Hd, Hd, Hd, Hd, Hd, 1, 1, 0,0,0,0,0,0);
        gemm(stream, 0, 0, h, kw, kbias, kb_, (int)M2, Hd, Hd, Hd, Hd, Hd, 1, 1, 0,0,0,0,0,0);
        gemm(stream, 0, 0, h, vw, vbias, vb_, (int)M2, Hd, Hd, Hd, Hd, Hd, 1, 1, 0,0,0,0,0,0);
        // scores[b,h,q,k] = Q . K^T   (batch = B*NH, B-side transposed)
        gemm(stream, 1, 0, qb_, kb_, nullptr, big, SYq, SYq, HDd, Hd, Hd, SYq,
             BB * NHh, NHh,
             (long)SYq * Hd, HDd, (long)SYq * Hd, HDd,
             (long)NHh * SYq * SYq, (long)SYq * SYq);
        attn_softmax_kernel<<<smBlocksS, b256, 0, stream>>>(big, y_mask, SYq, SYq, 1);
        // ctx[b,q,h*64+d] = P . V
        gemm(stream, 0, 0, big, vb_, nullptr, c1, SYq, HDd, SYq, SYq, Hd, Hd,
             BB * NHh, NHh,
             (long)NHh * SYq * SYq, (long)SYq * SYq, (long)SYq * Hd, HDd,
             (long)SYq * Hd, HDd);
        gemm(stream, 0, 0, c1, ow, obias, c2, (int)M2, Hd, Hd, Hd, Hd, Hd, 1, 1, 0,0,0,0,0,0);
        ln_kernel<<<lnBlocks, b256, 0, stream>>>(c2, h, l1s, l1b, h, (int)M2);

        // --- cross attention ---
        gemm(stream, 0, 0, h,      cqw, cqb, qb_, (int)M2, Hd, Hd, Hd, Hd, Hd, 1, 1, 0,0,0,0,0,0);
        gemm(stream, 0, 0, x_reps, ckw, ckb, kb_, (int)MX, Hd, Hd, Hd, Hd, Hd, 1, 1, 0,0,0,0,0,0);
        gemm(stream, 0, 0, x_reps, cvw, cvb, vb_, (int)MX, Hd, Hd, Hd, Hd, Hd, 1, 1, 0,0,0,0,0,0);
        gemm(stream, 1, 0, qb_, kb_, nullptr, big, SYq, SXk, HDd, Hd, Hd, SXk,
             BB * NHh, NHh,
             (long)SYq * Hd, HDd, (long)SXk * Hd, HDd,
             (long)NHh * SYq * SXk, (long)SYq * SXk);
        attn_softmax_kernel<<<smBlocksS, b256, 0, stream>>>(big, x_mask, SYq, SXk, 0);
        gemm(stream, 0, 0, big, vb_, nullptr, c1, SYq, HDd, SXk, SXk, Hd, Hd,
             BB * NHh, NHh,
             (long)NHh * SYq * SXk, (long)SYq * SXk, (long)SXk * Hd, HDd,
             (long)SYq * Hd, HDd);
        gemm(stream, 0, 0, c1, cow, cob, c2, (int)M2, Hd, Hd, Hd, Hd, Hd, 1, 1, 0,0,0,0,0,0);
        ln_kernel<<<lnBlocks, b256, 0, stream>>>(c2, h, l2s, l2b, h, (int)M2);

        // --- FFN ---
        gemm(stream, 0, 1, h, f1w, f1b, big, (int)M2, FFd, Hd, Hd, FFd, FFd, 1, 1, 0,0,0,0,0,0);
        gemm(stream, 0, 0, big, f2w, f2b, c2, (int)M2, Hd, FFd, FFd, Hd, Hd, 1, 1, 0,0,0,0,0,0);
        ln_kernel<<<lnBlocks, b256, 0, stream>>>(c2, h, l3s, l3b, h, (int)M2);
    }
    // h now holds y_rep

    // ---- LM head: dense -> gelu -> LN -> decoder -> log_softmax ----
    gemm(stream, 0, 1, h, F(34), F(35), c1, (int)M2, Hd, Hd, Hd, Hd, Hd, 1, 1, 0,0,0,0,0,0);
    ln_kernel<<<lnBlocks, b256, 0, stream>>>(c1, nullptr, F(36), F(37), c2, (int)M2);
    gemm(stream, 0, 0, c2, F(38), F(39), out_lm, (int)M2, Vv, Hd, Hd, Vv, Vv, 1, 1, 0,0,0,0,0,0);
    logsoftmax_kernel<<<(int)M2, b256, 0, stream>>>(out_lm, Vv);

    // ---- span heads via highway nets (pad_bias cancels under log_softmax) ----
    const int hwIdx[2] = {40, 44};
    float* spanOut[2] = {out_sb, out_se};
    for (int hsel = 0; hsel < 2; ++hsel) {
        hipMemcpyAsync(hy, h, (size_t)M2 * Hd * sizeof(float),
                       hipMemcpyDeviceToDevice, stream);
        const float* gw = F(hwIdx[hsel] + 0); const float* gb = F(hwIdx[hsel] + 1);
        const float* nw = F(hwIdx[hsel] + 2); const float* nb = F(hwIdx[hsel] + 3);
        for (int n = 0; n < NHWc; ++n) {
            gemm(stream, 0, 2, hy, gw + (size_t)n * HH, gb + (size_t)n * Hd, c2,
                 (int)M2, Hd, Hd, Hd, Hd, Hd, 1, 1, 0,0,0,0,0,0);
            gemm(stream, 0, 3, hy, nw + (size_t)n * HH, nb + (size_t)n * Hd, c3,
                 (int)M2, Hd, Hd, Hd, Hd, Hd, 1, 1, 0,0,0,0,0,0);
            hw_combine_kernel<<<elemBlocks, b256, 0, stream>>>(c2, c3, hy);
        }
        // span[b,y,x] = hy . x_reps^T  (batched over B)
        gemm(stream, 1, 0, hy, x_reps, nullptr, spanOut[hsel], SYq, SXk, Hd, Hd, Hd, SXk,
             BB, 1,
             (long)SYq * Hd, 0, (long)SXk * Hd, 0, (long)SYq * SXk, 0);
        logsoftmax_kernel<<<(int)M2, b256, 0, stream>>>(spanOut[hsel], SXk);
    }

    // ---- ts heads: (y_rep @ W + b) . x_reps^T ----
    const int tsW[2] = {48, 50};
    float* tsOut[2] = {out_ts, out_te};
    for (int hsel = 0; hsel < 2; ++hsel) {
        gemm(stream, 0, 0, h, F(tsW[hsel]), F(tsW[hsel] + 1), c2,
             (int)M2, Hd, Hd, Hd, Hd, Hd, 1, 1, 0,0,0,0,0,0);
        gemm(stream, 1, 0, c2, x_reps, nullptr, tsOut[hsel], SYq, SXk, Hd, Hd, Hd, SXk,
             BB, 1,
             (long)SYq * Hd, 0, (long)SXk * Hd, 0, (long)SYq * SXk, 0);
        logsoftmax_kernel<<<(int)M2, b256, 0, stream>>>(tsOut[hsel], SXk);
    }
}
